// ConvTSP_32633161515437
// MI455X (gfx1250) — matrix operand order repaired and checked by reading, not compiled, not run on hardware
//
#include <hip/hip_runtime.h>
#include <cstdint>
#include <cstddef>

#define HDIM 128
#define NN 200
#define BB 4
#define LL 3
#define EDGES (BB*NN*NN)   /* 160000 */
#define NODES (BB*NN)      /* 800 */

typedef __attribute__((ext_vector_type(16))) __bf16 v16bf;
typedef __attribute__((ext_vector_type(8)))  float  v8f;

static __device__ __forceinline__ unsigned short f2bf(float f){
    unsigned int u = __float_as_uint(f);
    unsigned int r = u + 0x7FFFu + ((u >> 16) & 1u);   // RNE
    return (unsigned short)(r >> 16);
}
static __device__ __forceinline__ unsigned int pack2bf(float a, float b){
    return (unsigned int)f2bf(a) | ((unsigned int)f2bf(b) << 16);
}

// ---------- setup kernels ----------------------------------------------------

__global__ void k_mask_sums(const float* __restrict__ mask, float* __restrict__ st){
    __shared__ float red[256];
    int t = threadIdx.x;
    float cn = 0.f, ce = 0.f;
    for (int b = 0; b < BB; ++b){
        float v = 0.f;
        for (int n = t; n < NN; n += 256) v += mask[b*NN + n];
        red[t] = v; __syncthreads();
        for (int s = 128; s > 0; s >>= 1){ if (t < s) red[t] += red[t+s]; __syncthreads(); }
        if (t == 0){ cn += red[0]; ce += red[0]*red[0]; }
        __syncthreads();
    }
    if (t == 0){ st[0] = cn; st[1] = ce; }
}

// Pack W (K x N, row major f32) into per-lane WMMA B fragments (bf16 pairs).
// Fragment layout: lane holds column n = lane%16; VGPR v -> K pair (8*(lane/16) + 2*(v%4) + 16*(v/4), +1).
__global__ void k_pack_w(const float* __restrict__ eUw, const float* __restrict__ mlpUw,
                         unsigned int* __restrict__ Wpk){
    int g = blockIdx.x*blockDim.x + threadIdx.x;   // 4 matrices * 8192 uints
    if (g >= 4*8192) return;
    int mat  = g >> 13;
    int rem  = g & 8191;
    int ks   = rem >> 11;          // k-step (32 wide)
    int rem2 = rem & 2047;
    int nt   = rem2 >> 8;          // n tile (16 wide)
    int rem3 = rem2 & 255;
    int lane = rem3 >> 3;
    int v    = rem3 & 7;
    const float* W = (mat < 3) ? (eUw + (size_t)mat*HDIM*HDIM) : mlpUw;
    int n = nt*16 + (lane & 15);
    int k = ks*32 + 8*(lane >> 4) + 2*(v & 3) + 16*(v >> 2);
    Wpk[g] = pack2bf(W[k*HDIM + n], W[(k+1)*HDIM + n]);
}

__global__ void k_embed_nodes(const float* __restrict__ coord, const float* __restrict__ wc,
                              float* __restrict__ x){
    int p = blockIdx.x, t = threadIdx.x;
    float c0 = coord[p*2+0], c1 = coord[p*2+1];
    x[p*HDIM + t] = c0*wc[t] + c1*wc[HDIM + t];
}

__global__ void k_embed_edges(const int* __restrict__ xe, const float* __restrict__ ev,
                              const float* __restrict__ weval, const float* __restrict__ emb,
                              float* __restrict__ E){
    int e = blockIdx.x, t = threadIdx.x;
    float v = (t < 64) ? ev[e]*weval[t] : emb[xe[e]*64 + (t-64)];
    E[(size_t)e*HDIM + t] = v;
}

// ---------- per-layer kernels ------------------------------------------------

// Vx_e = m*(x@eV_w + b) ; Ux = m*(x@nU_w + b) ; Vx = m*(x@nV_w + b). 800 rows -> plain VALU.
__global__ void k_node_gemm(const float* __restrict__ x, const float* __restrict__ mask,
                            const float* __restrict__ eVw, const float* __restrict__ eVb,
                            const float* __restrict__ nUw, const float* __restrict__ nUb,
                            const float* __restrict__ nVw, const float* __restrict__ nVb,
                            float* __restrict__ Vxe, float* __restrict__ Ux,
                            float* __restrict__ Vx){
    __shared__ float xr[HDIM];
    int p = blockIdx.x, t = threadIdx.x;
    xr[t] = x[p*HDIM + t];
    __syncthreads();
    float ae = 0.f, au = 0.f, av = 0.f;
    #pragma unroll 4
    for (int k = 0; k < HDIM; ++k){
        float xk = xr[k];
        ae += xk * eVw[k*HDIM + t];
        au += xk * nUw[k*HDIM + t];
        av += xk * nVw[k*HDIM + t];
    }
    float m = mask[p];
    Vxe[p*HDIM+t] = m*(ae + eVb[t]);
    Ux [p*HDIM+t] = m*(au + nUb[t]);
    Vx [p*HDIM+t] = m*(av + nVb[t]);
}

// WMMA edge GEMM: out[row,:] = E[row,:] @ W (+ epilogue). 64 rows per WG, 8 waves.
// mode 0: += bias + Vxe[b,i] + Vxe[b,j]   (e_tmp)
// mode 1: relu(+bias)                      (MLP hidden)
__global__ __launch_bounds__(256) void k_edge_gemm(
        const float* __restrict__ E, const unsigned int* __restrict__ Wpk,
        const float* __restrict__ bias, const float* __restrict__ Vxe,
        float* __restrict__ out, int mode){
    __shared__ unsigned short eT[64*132];   // bf16 tile, rows padded to 132 (bank-conflict free)
    const int tid   = threadIdx.x;
    const int rBase = blockIdx.x * 64;

    // stage 64x128 f32 -> bf16 in LDS (coalesced float4 loads)
    const float4* src = (const float4*)(E + (size_t)rBase*HDIM);
    unsigned int* eTu = (unsigned int*)eT;
    #pragma unroll
    for (int it = 0; it < 8; ++it){
        int idx4 = tid + it*256;            // 0..2047
        int row  = idx4 >> 5;
        int kf   = (idx4 & 31) * 4;
        float4 f = src[idx4];
        int o = (row*132 + kf) >> 1;
        eTu[o+0] = pack2bf(f.x, f.y);
        eTu[o+1] = pack2bf(f.z, f.w);
    }
    __syncthreads();

    const int lane  = tid & 31;
    const int wave  = tid >> 5;
    const int mRow  = (wave & 3) * 16;      // 4 M sub-tiles
    const int nBase = (wave >> 2) * 64;     // 2 N halves, 4 tiles each

    v8f acc[4];
    #pragma unroll
    for (int t = 0; t < 4; ++t)
        #pragma unroll
        for (int q = 0; q < 8; ++q) acc[t][q] = 0.f;

    // A fragment: lane holds row M=lane%16; K pattern per ISA 16-bit A layout.
    const int arow = mRow + (lane & 15);
    const unsigned short* abase = eT + arow*132 + 8*(lane >> 4);

    #pragma unroll
    for (int ks = 0; ks < 4; ++ks){
        union { unsigned int u[8]; v16bf v; } A;
        #pragma unroll
        for (int v2 = 0; v2 < 8; ++v2){
            int kb = ks*32 + ((v2 & 3)*2) + ((v2 >> 2)*16);
            A.u[v2] = *(const unsigned int*)(abase + kb);
        }
        #pragma unroll
        for (int t = 0; t < 4; ++t){
            int ntile = (nBase >> 4) + t;
            const uint4* bp = (const uint4*)(Wpk + ((size_t)(ks*8 + ntile)*32 + lane)*8);
            union { uint4 q[2]; v16bf v; } B;
            B.q[0] = bp[0]; B.q[1] = bp[1];
            acc[t] = __builtin_amdgcn_wmma_f32_16x16x32_bf16(
                         false, A.v, false, B.v, (short)0, acc[t], false, false);
        }
    }

    // epilogue (C layout: VGPR r, lanes 0-15 -> M=r, lanes 16-31 -> M=r+8; N=lane%16)
    #pragma unroll
    for (int t = 0; t < 4; ++t){
        #pragma unroll
        for (int r = 0; r < 8; ++r){
            int row = rBase + mRow + r + 8*(lane >> 4);
            int c   = nBase + t*16 + (lane & 15);
            float v = acc[t][r];
            if (mode == 0){
                int b   = row / (NN*NN);
                int rem = row - b*NN*NN;
                int i   = rem / NN;
                int j   = rem - i*NN;
                v += bias[c] + Vxe[(b*NN + i)*HDIM + c] + Vxe[(b*NN + j)*HDIM + c];
            } else {
                v = fmaxf(v + bias[c], 0.f);
            }
            out[(size_t)row*HDIM + c] = v;
        }
    }
}

// Per (b,i): gate = sigmoid(e_tmp)*sq_mask; agg/denom over j; x_tmp; BN partial sums.
__global__ void k_edge_reduce(const float* __restrict__ Etmp, const float* __restrict__ mask,
                              const float* __restrict__ Vx, const float* __restrict__ Ux,
                              float* __restrict__ Xtmp, float* __restrict__ part){
    __shared__ float mj_s[NN];
    int p = blockIdx.x, t = threadIdx.x;
    int b = p / NN;
    for (int j = t; j < NN; j += HDIM) mj_s[j] = mask[b*NN + j];
    __syncthreads();
    float mi = mask[p];
    float agg = 0.f, den = 0.f, sE = 0.f, sE2 = 0.f;
    const float* er  = Etmp + (size_t)p*NN*HDIM;
    const float* vxr = Vx   + (size_t)b*NN*HDIM;
    for (int j = 0; j < NN; ++j){
        float et = er[j*HDIM + t];
        float sm = mi * mj_s[j];
        float g  = sm / (1.f + __expf(-et));
        agg += g * vxr[j*HDIM + t];
        den += g;
        sE  += et * sm;
        sE2 += et * et * sm;
    }
    float xt = Ux[p*HDIM + t] + agg / (1e-20f + den);
    Xtmp[p*HDIM + t] = xt;
    part[((size_t)0*NODES + p)*HDIM + t] = sE;
    part[((size_t)1*NODES + p)*HDIM + t] = sE2;
    part[((size_t)2*NODES + p)*HDIM + t] = mi*xt;
    part[((size_t)3*NODES + p)*HDIM + t] = mi*xt*xt;
}

// Deterministic stat finalize: coefs = {aE, bE, aN, bN} with a = g*rsqrt(var+eps), b = beta - mean*a.
__global__ void k_finalize(const float* __restrict__ st, const float* __restrict__ part,
                           const float* __restrict__ gE, const float* __restrict__ bEp,
                           const float* __restrict__ gN, const float* __restrict__ bNp,
                           float* __restrict__ coefs){
    int t = threadIdx.x;
    float sE = 0.f, sE2 = 0.f, sN = 0.f, sN2 = 0.f;
    for (int p = 0; p < NODES; ++p){
        sE  += part[((size_t)0*NODES + p)*HDIM + t];
        sE2 += part[((size_t)1*NODES + p)*HDIM + t];
        sN  += part[((size_t)2*NODES + p)*HDIM + t];
        sN2 += part[((size_t)3*NODES + p)*HDIM + t];
    }
    float cntN = st[0], cntE = st[1];
    float mE = sE / cntE;
    float vE = sE2 / cntE - mE*mE;
    float aE = gE[t] * rsqrtf(vE + 1e-5f);
    coefs[t]            = aE;
    coefs[HDIM + t]     = bEp[t] - mE*aE;
    float mN = sN / cntN;
    float vN = sN2 / cntN - mN*mN;
    float aN = gN[t] * rsqrtf(vN + 1e-5f);
    coefs[2*HDIM + t]   = aN;
    coefs[3*HDIM + t]   = bNp[t] - mN*aN;
}

__global__ void k_node_update(const float* __restrict__ Xtmp, const float* __restrict__ mask,
                              const float* __restrict__ coefs, float* __restrict__ x){
    int p = blockIdx.x, t = threadIdx.x;
    float xt = Xtmp[p*HDIM + t];
    float xn = (mask[p] > 0.f) ? (coefs[2*HDIM+t]*xt + coefs[3*HDIM+t]) : xt;
    x[p*HDIM + t] += fmaxf(xn, 0.f);
}

__global__ void k_edge_update(const float* __restrict__ Etmp, const float* __restrict__ mask,
                              const float* __restrict__ coefs, float* __restrict__ E){
    int e = blockIdx.x, t = threadIdx.x;
    int b   = e / (NN*NN);
    int rem = e - b*NN*NN;
    int i   = rem / NN;
    int j   = rem - i*NN;
    float sm = mask[b*NN + i] * mask[b*NN + j];
    float et = Etmp[(size_t)e*HDIM + t];
    float en = (sm > 0.f) ? (coefs[t]*et + coefs[HDIM+t]) : et;
    E[(size_t)e*HDIM + t] += fmaxf(en, 0.f);
}

// y[e, 0:2] = h[e,:] @ mlp_V + b  — one wave per edge, shuffle reduction.
__global__ void k_mlp_out(const float* __restrict__ Hbuf, const float* __restrict__ Vw,
                          const float* __restrict__ Vb, float* __restrict__ y){
    int gw   = blockIdx.x * (blockDim.x >> 5) + (threadIdx.x >> 5);
    int lane = threadIdx.x & 31;
    if (gw >= EDGES) return;
    const float* hr = Hbuf + (size_t)gw*HDIM;
    float a0 = 0.f, a1 = 0.f;
    #pragma unroll
    for (int q = 0; q < 4; ++q){
        int hh = q*32 + lane;
        float v = hr[hh];
        a0 += v * Vw[hh*2 + 0];
        a1 += v * Vw[hh*2 + 1];
    }
    #pragma unroll
    for (int off = 16; off >= 1; off >>= 1){
        a0 += __shfl_xor(a0, off, 32);
        a1 += __shfl_xor(a1, off, 32);
    }
    if (lane == 0){ y[gw*2 + 0] = a0 + Vb[0]; y[gw*2 + 1] = a1 + Vb[1]; }
}

// ---------- driver -----------------------------------------------------------

extern "C" void kernel_launch(void* const* d_in, const int* in_sizes, int n_in,
                              void* d_out, int out_size, void* d_ws, size_t ws_size,
                              hipStream_t stream){
    (void)in_sizes; (void)n_in; (void)out_size;
    const int*   xe    = (const int*)  d_in[0];
    const float* ev    = (const float*)d_in[1];
    const float* coord = (const float*)d_in[2];
    const float* mask  = (const float*)d_in[3];
    const float* wc    = (const float*)d_in[4];
    const float* weval = (const float*)d_in[5];
    const float* emb   = (const float*)d_in[6];
    const float* eUw   = (const float*)d_in[7];
    const float* eUb   = (const float*)d_in[8];
    const float* eVw   = (const float*)d_in[9];
    const float* eVb   = (const float*)d_in[10];
    const float* nUw   = (const float*)d_in[11];
    const float* nUb   = (const float*)d_in[12];
    const float* nVw   = (const float*)d_in[13];
    const float* nVb   = (const float*)d_in[14];
    const float* bnEg  = (const float*)d_in[15];
    const float* bnEb  = (const float*)d_in[16];
    const float* bnNg  = (const float*)d_in[17];
    const float* bnNb  = (const float*)d_in[18];
    const float* mUw   = (const float*)d_in[19];
    const float* mUb   = (const float*)d_in[20];
    const float* mVw   = (const float*)d_in[21];
    const float* mVb   = (const float*)d_in[22];
    float* out = (float*)d_out;

    char*  w   = (char*)d_ws;
    size_t off = 0;
    auto alloc = [&](size_t bytes)->char*{
        char* p = w + off; off += (bytes + 255) & ~(size_t)255; return p;
    };
    float*        EB  = (float*)alloc((size_t)EDGES*HDIM*4);      // e residual stream
    float*        ET  = (float*)alloc((size_t)EDGES*HDIM*4);      // e_tmp / MLP hidden
    float*        XB  = (float*)alloc((size_t)NODES*HDIM*4);      // x residual stream
    float*        XT  = (float*)alloc((size_t)NODES*HDIM*4);      // x_tmp
    float*        VXE = (float*)alloc((size_t)NODES*HDIM*4);
    float*        UXB = (float*)alloc((size_t)NODES*HDIM*4);
    float*        VXB = (float*)alloc((size_t)NODES*HDIM*4);
    unsigned int* WPK = (unsigned int*)alloc((size_t)4*8192*4);   // packed bf16 B fragments
    float*        PRT = (float*)alloc((size_t)4*NODES*HDIM*4);    // BN partial sums
    float*        ST  = (float*)alloc(4096);                      // [0]=cntN [1]=cntE
    float*        CF  = ST + 64;                                  // coefs aE,bE,aN,bN
    if (ws_size < off) return;   // workspace too small (needs ~168 MB)

    k_mask_sums  <<<1, 256, 0, stream>>>(mask, ST);
    k_pack_w     <<<128, 256, 0, stream>>>(eUw, mUw, WPK);
    k_embed_nodes<<<NODES, HDIM, 0, stream>>>(coord, wc, XB);
    k_embed_edges<<<EDGES, HDIM, 0, stream>>>(xe, ev, weval, emb, EB);

    for (int i = 0; i < LL; ++i){
        k_node_gemm  <<<NODES, HDIM, 0, stream>>>(XB, mask,
            eVw + (size_t)i*HDIM*HDIM, eVb + i*HDIM,
            nUw + (size_t)i*HDIM*HDIM, nUb + i*HDIM,
            nVw + (size_t)i*HDIM*HDIM, nVb + i*HDIM,
            VXE, UXB, VXB);
        k_edge_gemm  <<<EDGES/64, 256, 0, stream>>>(EB, WPK + (size_t)i*8192,
            eUb + i*HDIM, VXE, ET, 0);
        k_edge_reduce<<<NODES, HDIM, 0, stream>>>(ET, mask, VXB, UXB, XT, PRT);
        k_finalize   <<<1, HDIM, 0, stream>>>(ST, PRT,
            bnEg + i*HDIM, bnEb + i*HDIM, bnNg + i*HDIM, bnNb + i*HDIM, CF);
        k_node_update<<<NODES, HDIM, 0, stream>>>(XT, mask, CF, XB);
        k_edge_update<<<EDGES, HDIM, 0, stream>>>(ET, mask, CF, EB);
    }

    // MLP head: h = relu(e @ mlp_U + b) -> ET ; y = h @ mlp_V + b -> out
    k_edge_gemm<<<EDGES/64, 256, 0, stream>>>(EB, WPK + (size_t)3*8192, mUb, nullptr, ET, 1);
    k_mlp_out  <<<(EDGES + 7)/8, 256, 0, stream>>>(ET, mVw, mVb, out);
}
